// STAttention_86277303042069
// MI455X (gfx1250) — matrix operand order repaired and checked by reading, not compiled, not run on hardware
//
#include <hip/hip_runtime.h>
#include <hip/hip_bf16.h>

typedef __bf16 bf16;
typedef bf16  v16bf  __attribute__((ext_vector_type(16)));
typedef bf16  bf16x8 __attribute__((ext_vector_type(8)));
typedef float v8f    __attribute__((ext_vector_type(8)));

// ln(10000)/16 : freq_t = exp(-t * ln(10000)/16)  (dh = 32, freqs = 10000^(-2t/32))
#define ROPE_LOGF 0.57564627324851148f

// ---- WMMA fragment loaders (16-bit, wave32) --------------------------------
// A (16x32, M x K): lane L -> m = L%16, element i -> k = (i/8)*16 + (L/16)*8 + i%8
static __device__ __forceinline__ v16bf load_a_frag(const bf16* row, int kb, int g) {
  bf16x8 a0 = *(const bf16x8*)(row + kb + g * 8);
  bf16x8 a1 = *(const bf16x8*)(row + kb + 16 + g * 8);
  v16bf af;
#pragma unroll
  for (int i = 0; i < 8; ++i) { af[i] = a0[i]; af[i + 8] = a1[i]; }
  return af;
}
// B (32x16, K x N), stored as BT[n][k]: lane L -> n = L%16, element i -> k = (i/8)*16 + (L/16)*8 + i%8
static __device__ __forceinline__ v16bf load_b_frag(const bf16* row, int kb, int g) {
  bf16x8 b0 = *(const bf16x8*)(row + kb + g * 8);
  bf16x8 b1 = *(const bf16x8*)(row + kb + 16 + g * 8);
  v16bf bf;
#pragma unroll
  for (int i = 0; i < 8; ++i) { bf[i] = b0[i]; bf[i + 8] = b1[i]; }
  return bf;
}

// ---- elementwise converts ---------------------------------------------------
__global__ void f32_to_bf16_kernel(const float* __restrict__ in, bf16* __restrict__ out, int n) {
  int i = blockIdx.x * 256 + threadIdx.x;
  if (i < n) out[i] = (bf16)in[i];
}

// W[k][n] f32 -> WT[n][k] bf16
__global__ void transpose_to_bf16_kernel(const float* __restrict__ W, bf16* __restrict__ WT,
                                         int K, int N) {
  int i = blockIdx.x * 256 + threadIdx.x;
  if (i < K * N) {
    int k = i / N, n = i % N;
    WT[(size_t)n * K + k] = (bf16)W[i];
  }
}

// ---- GEMM: C[M][N] = A[M][K](bf16) * BT[N][K](bf16), f32 out ---------------
// grid.x = M/16, grid.y*4 = N/32, 128 threads (4 waves), wave-per-16x32 C tile.
// Software-pipelined k-loop: next fragments fetched before current WMMA pair.
__global__ __launch_bounds__(128) void gemm_bf16_wmma(const bf16* __restrict__ A,
                                                      const bf16* __restrict__ BT,
                                                      float* __restrict__ C, int N, int K) {
  const int tid = threadIdx.x;
  const int wave = tid >> 5, lane = tid & 31;
  const int g = lane >> 4, ln = lane & 15;
  const int tm = blockIdx.x;
  const int tn = blockIdx.y * 4 + wave;

  const bf16* arow  = A  + (size_t)(tm * 16 + ln) * K;
  const bf16* brow0 = BT + (size_t)(tn * 32 + ln) * K;
  const bf16* brow1 = BT + (size_t)(tn * 32 + 16 + ln) * K;

  v8f acc0 = {}, acc1 = {};
  v16bf af = load_a_frag(arow, 0, g);
  v16bf b0 = load_b_frag(brow0, 0, g);
  v16bf b1 = load_b_frag(brow1, 0, g);
  for (int kb = 0; kb < K; kb += 32) {
    const int kn = (kb + 32 < K) ? kb + 32 : kb;
    v16bf afn = load_a_frag(arow, kn, g);
    v16bf b0n = load_b_frag(brow0, kn, g);
    v16bf b1n = load_b_frag(brow1, kn, g);
    if (kb + 64 < K) {
      __builtin_prefetch(arow + kb + 64, 0, 3);
      __builtin_prefetch(brow0 + kb + 64, 0, 3);
      __builtin_prefetch(brow1 + kb + 64, 0, 3);
    }
    acc0 = __builtin_amdgcn_wmma_f32_16x16x32_bf16(false, af, false, b0, (short)0, acc0,
                                                   false, false);
    acc1 = __builtin_amdgcn_wmma_f32_16x16x32_bf16(false, af, false, b1, (short)0, acc1,
                                                   false, false);
    af = afn; b0 = b0n; b1 = b1n;
  }
  float* crow = C + (size_t)(tm * 16) * N + tn * 32 + ln;
#pragma unroll
  for (int r = 0; r < 8; ++r) crow[(size_t)(r + 8 * g) * N] = acc0[r];
#pragma unroll
  for (int r = 0; r < 8; ++r) crow[(size_t)(r + 8 * g) * N + 16] = acc1[r];
}

// ---- RoPE + scatter into attention-tile layout -----------------------------
// Qproj: [16384][512] f32, row = b*4096 + y*64 + x, col = head*64 + d
// Qt: [b][h][qtile(64)][qi(64)][d(64)] bf16
__global__ void rope_q_kernel(const float* __restrict__ Qproj, const float* __restrict__ qc,
                              bf16* __restrict__ Qt) {
  int idx = blockIdx.x * 256 + threadIdx.x;  // pair index: 16384 * 256
  if (idx >= 16384 * 256) return;
  int m = idx >> 8, cp = idx & 255;
  int head = cp >> 5, ph = cp & 31;
  int b = m >> 12, n = m & 4095;
  float cx = qc[((size_t)b * 4096 + n) * 2 + 0];
  float cy = qc[((size_t)b * 4096 + n) * 2 + 1];
  int t, d0; float coord;
  if (ph < 16) { t = ph;      d0 = 2 * ph;        coord = cx; }
  else         { t = ph - 16; d0 = 32 + 2 * t;    coord = cy; }
  float ang = coord * __expf(-(float)t * ROPE_LOGF);
  float s, c; __sincosf(ang, &s, &c);
  const float* src = Qproj + (size_t)m * 512 + head * 64 + d0;
  float v0 = src[0], v1 = src[1];
  float o0 = v0 * c - v1 * s, o1 = v0 * s + v1 * c;
  int y = n >> 6, x = n & 63;
  int qtile = (y >> 3) * 8 + (x >> 3);
  int qi = (y & 7) * 8 + (x & 7);
  bf16* dst = Qt + (((size_t)(b * 8 + head) * 64 + qtile) * 4096 + qi * 64 + d0);
  dst[0] = (bf16)o0; dst[1] = (bf16)o1;
}

// KVproj: [65536][1024] f32, cols: [0,512)=K (rope), [512,1024)=V (copy)
// Kt: [b][h][ktile(256)][ki(64)][d(64)]  (row-major keys)
// Vt: [b][h][ktile(256)][d(64)][ki(64)]  (TRANSPOSED within tile: ready as WMMA B operand)
__global__ void rope_kv_kernel(const float* __restrict__ KVproj, const float* __restrict__ kc,
                               bf16* __restrict__ Kt, bf16* __restrict__ Vt) {
  int idx = blockIdx.x * 256 + threadIdx.x;  // pair index: 65536 * 512
  if (idx >= 65536 * 512) return;
  int m = idx >> 9, cp = idx & 511;
  int b = m >> 14, n = m & 16383;
  int y = n >> 7, x = n & 127;
  int ktile = (y >> 3) * 16 + (x >> 3);
  int ki = (y & 7) * 8 + (x & 7);
  if (cp < 256) {  // K path with RoPE
    int head = cp >> 5, ph = cp & 31;
    float cx = kc[((size_t)b * 16384 + n) * 2 + 0];
    float cy = kc[((size_t)b * 16384 + n) * 2 + 1];
    int t, d0; float coord;
    if (ph < 16) { t = ph;      d0 = 2 * ph;     coord = cx; }
    else         { t = ph - 16; d0 = 32 + 2 * t; coord = cy; }
    float ang = coord * __expf(-(float)t * ROPE_LOGF);
    float s, c; __sincosf(ang, &s, &c);
    const float* src = KVproj + (size_t)m * 1024 + head * 64 + d0;
    float v0 = src[0], v1 = src[1];
    bf16* dst = Kt + (((size_t)(b * 8 + head) * 256 + ktile) * 4096 + ki * 64 + d0);
    dst[0] = (bf16)(v0 * c - v1 * s);
    dst[1] = (bf16)(v0 * s + v1 * c);
  } else {  // V path: straight convert, transposed tile layout [d][ki]
    int cp2 = cp - 256;
    int head = cp2 >> 5, d0 = (cp2 & 31) * 2;
    const float* src = KVproj + (size_t)m * 1024 + 512 + head * 64 + d0;
    bf16* dst = Vt + (((size_t)(b * 8 + head) * 256 + ktile) * 4096 + (size_t)d0 * 64 + ki);
    dst[0]  = (bf16)src[0];
    dst[64] = (bf16)src[1];
  }
}

// ---- Sliding-tile attention: one block per (b, h, qtile) -------------------
// Double-buffered K/V tiles fetched with GLOBAL_LOAD_ASYNC_TO_LDS_B128 (ASYNCcnt),
// overlapping the next tile's fetch with the current tile's WMMA + softmax.
__global__ __launch_bounds__(128) void sta_attn_kernel(const bf16* __restrict__ Qt,
                                                       const bf16* __restrict__ Kt,
                                                       const bf16* __restrict__ Vt,
                                                       bf16* __restrict__ Obf) {
  const int qtile = blockIdx.x;  // 0..63
  const int h = blockIdx.y;      // 0..7
  const int b = blockIdx.z;      // 0..3
  const int tid = threadIdx.x;
  const int wave = tid >> 5, lane = tid & 31;
  const int g = lane >> 4, ln = lane & 15;

  __shared__ alignas(32) bf16 Qs[64][64];
  __shared__ alignas(32) bf16 Ks[2][64][64];   // [buf][key][d]
  __shared__ alignas(32) bf16 Vts[2][64][64];  // [buf][d][key] (pre-transposed in global)
  __shared__ alignas(32) bf16 Ps[64][64];
  __shared__ float Ss[64][64];
  __shared__ float Oacc[64][64];
  __shared__ float mrow[64];
  __shared__ float lrow[64];

  const size_t bh = (size_t)b * 8 + h;
  const bf16* qbase = Qt + (bh * 64 + qtile) * 4096;
  {
    const uint4* sp = (const uint4*)qbase;
    uint4* dp = (uint4*)&Qs[0][0];
    for (int i = tid; i < 512; i += 128) dp[i] = sp[i];
  }
  for (int i = tid; i < 4096; i += 128) Oacc[i >> 6][i & 63] = 0.f;
  if (tid < 64) { mrow[tid] = -3.0e38f; lrow[tid] = 0.f; }

  const int ty = qtile >> 3, tx = qtile & 7;
  int cx = (tx * 15 + 3) / 7; cx = cx < 1 ? 1 : (cx > 14 ? 14 : cx);
  int cy = (ty * 15 + 3) / 7; cy = cy < 1 ? 1 : (cy > 14 ? 14 : cy);

  const uint32_t ldsK[2] = {(uint32_t)(size_t)&Ks[0][0][0], (uint32_t)(size_t)&Ks[1][0][0]};
  const uint32_t ldsV[2] = {(uint32_t)(size_t)&Vts[0][0][0], (uint32_t)(size_t)&Vts[1][0][0]};

  // this wave's quarter (2 KB) of an 8 KB tile: 4 async b128 transfers
  auto issue_tile = [&](const bf16* gbase, uint32_t lbase) {
    uint32_t o = (uint32_t)(wave * 2048 + lane * 16);
#pragma unroll
    for (int i = 0; i < 4; ++i) {
      uint32_t go = o + (uint32_t)(i * 512);
      uint32_t lo = lbase + go;
      asm volatile("global_load_async_to_lds_b128 %0, %1, %2"
                   :: "v"(lo), "v"(go), "s"(gbase)
                   : "memory");
    }
  };
  auto tile_off = [&](int j) -> size_t {
    int kt = (cy + j / 3 - 1) * 16 + (cx + j % 3 - 1);
    return (bh * 256 + (size_t)kt) * 4096;
  };

  // prologue: start tile 0
  issue_tile(Kt + tile_off(0), ldsK[0]);
  issue_tile(Vt + tile_off(0), ldsV[0]);

  for (int j = 0; j < 9; ++j) {
    const int buf = j & 1;
    if (j + 1 < 9) {
      issue_tile(Kt + tile_off(j + 1), ldsK[buf ^ 1]);
      issue_tile(Vt + tile_off(j + 1), ldsV[buf ^ 1]);
      asm volatile("s_wait_asynccnt 0x8" ::: "memory");  // tile j landed, j+1 in flight
    } else {
      asm volatile("s_wait_asynccnt 0x0" ::: "memory");
    }
    __syncthreads();

    // S = (Q K^T) / 8 : wave handles 16 rows, loops 4 col-tiles
    for (int nt = 0; nt < 4; ++nt) {
      v8f acc = {};
      const bf16* arow = &Qs[wave * 16 + ln][0];
      const bf16* brow = &Ks[buf][nt * 16 + ln][0];
#pragma unroll
      for (int ks = 0; ks < 2; ++ks) {
        v16bf af = load_a_frag(arow, ks * 32, g);
        v16bf bfr = load_b_frag(brow, ks * 32, g);
        acc = __builtin_amdgcn_wmma_f32_16x16x32_bf16(false, af, false, bfr, (short)0, acc,
                                                      false, false);
      }
#pragma unroll
      for (int r = 0; r < 8; ++r)
        Ss[wave * 16 + r + 8 * g][nt * 16 + ln] = acc[r] * 0.125f;
    }
    __syncthreads();

    // online softmax: one thread per query row
    if (tid < 64) {
      const int q = tid;
      float mo = mrow[q], mx = mo;
      for (int cc = 0; cc < 64; ++cc) mx = fmaxf(mx, Ss[q][cc]);
      float scale = __expf(mo - mx);
      float l = lrow[q] * scale;
      for (int cc = 0; cc < 64; ++cc) {
        float p = __expf(Ss[q][cc] - mx);
        Ps[q][cc] = (bf16)p;
        l += p;
      }
      mrow[q] = mx; lrow[q] = l;
      for (int dd = 0; dd < 64; ++dd) Oacc[q][dd] *= scale;
    }
    __syncthreads();

    // Oacc += P @ V
    for (int nt = 0; nt < 4; ++nt) {
      v8f acc;
#pragma unroll
      for (int r = 0; r < 8; ++r) acc[r] = Oacc[wave * 16 + r + 8 * g][nt * 16 + ln];
      const bf16* arow = &Ps[wave * 16 + ln][0];
      const bf16* brow = &Vts[buf][nt * 16 + ln][0];
#pragma unroll
      for (int ks = 0; ks < 2; ++ks) {
        v16bf af = load_a_frag(arow, ks * 32, g);
        v16bf bfr = load_b_frag(brow, ks * 32, g);
        acc = __builtin_amdgcn_wmma_f32_16x16x32_bf16(false, af, false, bfr, (short)0, acc,
                                                      false, false);
      }
#pragma unroll
      for (int r = 0; r < 8; ++r) Oacc[wave * 16 + r + 8 * g][nt * 16 + ln] = acc[r];
    }
    __syncthreads();  // protects buf reuse at iteration j+2
  }

  // epilogue: normalize, write O in [b, y, x, head*64 + d] layout (bf16)
  const int yb = ty * 8, xb = tx * 8;
  for (int i = tid; i < 4096; i += 128) {
    int qi = i >> 6, dd = i & 63;
    int y = yb + (qi >> 3), x = xb + (qi & 7);
    size_t row = (size_t)b * 4096 + y * 64 + x;
    Obf[row * 512 + h * 64 + dd] = (bf16)(Oacc[qi][dd] / lrow[qi]);
  }
}

// ---- host-side orchestration -----------------------------------------------
extern "C" void kernel_launch(void* const* d_in, const int* in_sizes, int n_in,
                              void* d_out, int out_size, void* d_ws, size_t ws_size,
                              hipStream_t stream) {
  const float* tgt = (const float*)d_in[0];   // [4][64][64][512]
  const float* src = (const float*)d_in[1];   // [4][128][128][512]
  const float* qc  = (const float*)d_in[2];   // [4][4096][2]
  const float* kc  = (const float*)d_in[3];   // [4][16384][2]
  const float* Wq  = (const float*)d_in[4];   // [512][512]
  const float* Wkv = (const float*)d_in[5];   // [512][1024]
  const float* Wo  = (const float*)d_in[6];   // [512][512]
  float* out = (float*)d_out;                 // [4][64][64][512] f32

  char* w = (char*)d_ws;
  size_t off = 0;
  auto alloc = [&](size_t bytes) -> void* {
    void* p = w + off;
    off += (bytes + 255) & ~(size_t)255;
    return p;
  };
  bf16* tgtb   = (bf16*)alloc((size_t)16384 * 512 * 2);
  bf16* srcb   = (bf16*)alloc((size_t)65536 * 512 * 2);
  bf16* WqT    = (bf16*)alloc((size_t)512 * 512 * 2);
  bf16* WkvT   = (bf16*)alloc((size_t)1024 * 512 * 2);
  bf16* WoT    = (bf16*)alloc((size_t)512 * 512 * 2);
  float* Qproj = (float*)alloc((size_t)16384 * 512 * 4);
  float* KVproj= (float*)alloc((size_t)65536 * 1024 * 4);
  bf16* Qt     = (bf16*)alloc((size_t)4 * 8 * 64 * 4096 * 2);
  bf16* Kt     = (bf16*)alloc((size_t)4 * 8 * 256 * 4096 * 2);
  bf16* Vt     = (bf16*)alloc((size_t)4 * 8 * 256 * 4096 * 2);
  bf16* Obf    = (bf16*)alloc((size_t)16384 * 512 * 2);
  (void)ws_size; (void)in_sizes; (void)n_in; (void)out_size;

  // 1. precision converts
  f32_to_bf16_kernel<<<(16384 * 512) / 256, 256, 0, stream>>>(tgt, tgtb, 16384 * 512);
  f32_to_bf16_kernel<<<(65536 * 512) / 256, 256, 0, stream>>>(src, srcb, 65536 * 512);
  transpose_to_bf16_kernel<<<(512 * 512) / 256, 256, 0, stream>>>(Wq, WqT, 512, 512);
  transpose_to_bf16_kernel<<<(512 * 1024) / 256, 256, 0, stream>>>(Wkv, WkvT, 512, 1024);
  transpose_to_bf16_kernel<<<(512 * 512) / 256, 256, 0, stream>>>(Wo, WoT, 512, 512);

  // 2. projections (WMMA bf16, f32 accum); wave computes 16x32 C
  gemm_bf16_wmma<<<dim3(1024, 4), 128, 0, stream>>>(tgtb, WqT, Qproj, 512, 512);
  gemm_bf16_wmma<<<dim3(4096, 8), 128, 0, stream>>>(srcb, WkvT, KVproj, 1024, 512);

  // 3. RoPE + tile-layout scatter (V stored pre-transposed per tile)
  rope_q_kernel<<<(16384 * 256) / 256, 256, 0, stream>>>(Qproj, qc, Qt);
  rope_kv_kernel<<<(65536 * 512) / 256, 256, 0, stream>>>(KVproj, kc, Kt, Vt);

  // 4. sliding-tile attention (async double-buffered K/V in LDS)
  sta_attn_kernel<<<dim3(64, 8, 4), 128, 0, stream>>>(Qt, Kt, Vt, Obf);

  // 5. output projection -> f32 d_out
  gemm_bf16_wmma<<<dim3(1024, 4), 128, 0, stream>>>(Obf, WoT, out, 512, 512);
}
